// PVConv_24764781429165
// MI455X (gfx1250) — matrix operand order repaired and checked by reading, not compiled, not run on hardware
//
#include <hip/hip_runtime.h>

#define BB 8
#define CC 32
#define NN 16384
#define RR 32
#define VV 32768   // RR^3
#define CDIM 256
#define GC 4       // channels per group (G=8)

typedef __attribute__((ext_vector_type(16))) _Float16 v16h;
typedef __attribute__((ext_vector_type(8)))  float    v8f;

__device__ __forceinline__ float sigmoidf(float x){ return 1.f/(1.f+__expf(-x)); }

union H2U { _Float16 h[2]; unsigned int u; };

// ---------------- weight packing into WMMA A-fragment layout ----------------
// A-frag (16x32 f16, contiguous k order used here, not the ISA table order): lane l: M=l%16, khalf=l>>4;
// VGPR v holds K pair k0=khalf*16 + v*2.
__global__ void k_prep_w(const float* __restrict__ w1, const float* __restrict__ w2,
                         const float* __restrict__ pw,
                         unsigned int* wA1, unsigned int* wA2, unsigned int* wAp){
  int id = blockIdx.x*256 + threadIdx.x;
  if (id < 27648){                       // two convs, 13824 dwords each
    const float* w = (id < 13824) ? w1 : w2;
    unsigned int* dst = (id < 13824) ? wA1 : wA2;
    int e = (id < 13824) ? id : id - 13824;
    int v = e & 7, lane = (e>>3)&31, half = (e>>8)&1, tap = e>>9;
    int m = lane & 15, khalf = lane>>4;
    int k0 = (khalf<<4) + (v<<1);
    int co = half*16 + m;
    H2U p;
    p.h[0] = (_Float16)w[(co*CC + k0    )*27 + tap];
    p.h[1] = (_Float16)w[(co*CC + k0 + 1)*27 + tap];
    dst[e] = p.u;
  } else if (id < 28160){                // pc_w (32x32), single tap
    int e = id - 27648;
    int v = e & 7, lane = (e>>3)&31, half = (e>>8)&1;
    int m = lane & 15, khalf = lane>>4;
    int k0 = (khalf<<4) + (v<<1);
    int co = half*16 + m;
    H2U p;
    p.h[0] = (_Float16)pw[co*CC + k0];
    p.h[1] = (_Float16)pw[co*CC + k0 + 1];
    wAp[e] = p.u;
  }
}

// ---------------- channel-major f32 -> channel-last f16 staging ----------------
// xt[b][s][ci] so a WMMA B-fragment (16 consecutive ci) is one 32B load.
__global__ void k_tohalf(const float* __restrict__ x, _Float16* __restrict__ xt, int S){
  int bpb = S >> 8;                       // blocks per batch
  int b  = blockIdx.x / bpb;
  int s0 = (blockIdx.x % bpb) << 8;
  int tid = threadIdx.x;
  __shared__ _Float16 lds[256*33];        // [pos][ci], padded stride
  const float* xb = x + b*CC*S + s0;
  #pragma unroll
  for (int j=0; j<CC; ++j)
    lds[tid*33 + j] = (_Float16)0;        // placeholder to keep compiler happy (overwritten)
  for (int e = tid; e < 256*CC; e += 256){
    int ci = e >> 8, vx = e & 255;
    lds[vx*33 + ci] = (_Float16)xb[ci*S + vx];
  }
  __syncthreads();
  _Float16* ob = xt + ((size_t)b*S + s0 + tid)*CC;
  v16h lo, hi;
  #pragma unroll
  for (int k=0;k<16;++k){ lo[k] = lds[tid*33+k]; hi[k] = lds[tid*33+16+k]; }
  *(v16h*)ob = lo;
  *((v16h*)ob + 1) = hi;
}

// ---------------- coord normalization + voxel index ----------------
__global__ void k_coord(const float* __restrict__ coords, float* __restrict__ nc,
                        int* __restrict__ vidx){
  int b = blockIdx.x, tid = threadIdx.x;
  __shared__ float r0[256], r1[256], r2[256];
  __shared__ float sm0, sm1, sm2, sds;
  const float* c0 = coords + (b*3+0)*NN;
  const float* c1 = coords + (b*3+1)*NN;
  const float* c2 = coords + (b*3+2)*NN;
  float s0=0.f, s1=0.f, s2=0.f;
  for (int i=tid;i<NN;i+=256){ s0+=c0[i]; s1+=c1[i]; s2+=c2[i]; }
  r0[tid]=s0; r1[tid]=s1; r2[tid]=s2; __syncthreads();
  for (int st=128; st>0; st>>=1){
    if (tid<st){ r0[tid]+=r0[tid+st]; r1[tid]+=r1[tid+st]; r2[tid]+=r2[tid+st]; }
    __syncthreads();
  }
  if (tid==0){ sm0=r0[0]/NN; sm1=r1[0]/NN; sm2=r2[0]/NN; }
  __syncthreads();
  float m0=sm0, m1=sm1, m2=sm2;
  float mx=0.f;
  for (int i=tid;i<NN;i+=256){
    float dx=c0[i]-m0, dy=c1[i]-m1, dz=c2[i]-m2;
    mx = fmaxf(mx, sqrtf(dx*dx+dy*dy+dz*dz));
  }
  r0[tid]=mx; __syncthreads();
  for (int st=128; st>0; st>>=1){
    if (tid<st) r0[tid]=fmaxf(r0[tid],r0[tid+st]);
    __syncthreads();
  }
  if (tid==0) sds = 1.f/(2.f*fmaxf(r0[0],1e-12f));
  __syncthreads();
  float ds = sds;
  for (int i=tid;i<NN;i+=256){
    float nx = ((c0[i]-m0)*ds + 0.5f)*RR;
    float ny = ((c1[i]-m1)*ds + 0.5f)*RR;
    float nz = ((c2[i]-m2)*ds + 0.5f)*RR;
    nx = fminf(fmaxf(nx,0.f), RR-1.f);
    ny = fminf(fmaxf(ny,0.f), RR-1.f);
    nz = fminf(fmaxf(nz,0.f), RR-1.f);
    nc[(b*3+0)*NN+i]=nx; nc[(b*3+1)*NN+i]=ny; nc[(b*3+2)*NN+i]=nz;
    int vx=__float2int_rn(nx), vy=__float2int_rn(ny), vz=__float2int_rn(nz);
    vidx[b*NN+i] = (vx*RR+vy)*RR+vz;
  }
}

__global__ void k_zero(float* p, long count){
  long i = (long)blockIdx.x*256 + threadIdx.x;
  long stride = (long)gridDim.x*256;
  for (; i<count; i+=stride) p[i] = 0.f;
}

__global__ void k_scatter(const float* __restrict__ feat, const int* __restrict__ vidx,
                          float* __restrict__ vox, float* __restrict__ cnt){
  int i = blockIdx.x*256 + threadIdx.x;           // over B*C*N
  int n = i % NN;
  int c = (i / NN) % CC;
  int b = i / (CC*NN);
  int vid = vidx[b*NN+n];
  atomicAdd(&vox[(b*CC+c)*VV + vid], feat[i]);
  if (c == 0) atomicAdd(&cnt[b*VV + vid], 1.f);
}

__global__ void k_divide(float* __restrict__ vox, const float* __restrict__ cnt){
  int i = blockIdx.x*256 + threadIdx.x;           // over B*C*V
  int vid = i % VV;
  int b = i / (CC*VV);
  vox[i] /= fmaxf(cnt[b*VV+vid], 1.f);
}

// ---------------- modulation: m = silu(cond) @ modW^T + modB, for 3 mats ----------------
__global__ void k_mod(const float* __restrict__ cond,
                      const float* __restrict__ w1, const float* __restrict__ b1,
                      const float* __restrict__ w2, const float* __restrict__ b2,
                      const float* __restrict__ wp, const float* __restrict__ bp,
                      float* __restrict__ m){
  int b = blockIdx.x, tid = threadIdx.x;
  __shared__ float sc[CDIM];
  float x = cond[b*CDIM + tid];
  sc[tid] = x * sigmoidf(x);
  __syncthreads();
  if (tid < 192){
    int which = tid/64, o = tid%64;
    const float* w  = which==0 ? w1 : (which==1 ? w2 : wp);
    const float* bb = which==0 ? b1 : (which==1 ? b2 : bp);
    float acc = bb[o];
    for (int k=0;k<CDIM;++k) acc += sc[k]*w[o*CDIM+k];
    m[(which*BB + b)*64 + o] = acc;
  }
}

// ---------------- 3x3x3 conv as implicit GEMM on v_wmma_f32_16x16x32_f16 ----------------
// wave tile: 32 output channels x 16 voxels (contiguous in z), K=32 ci per tap.
// B-frag = single 32B contiguous load from channel-last f16 buffer; bounds via clamp+mask.
__global__ void k_conv(const _Float16* __restrict__ xt, const unsigned int* __restrict__ wA,
                       const float* __restrict__ bias, float* __restrict__ out){
  int lane = threadIdx.x & 31, wave = threadIdx.x >> 5;
  int tile = blockIdx.x*8 + wave;                 // 16384 tiles = B * V/16
  int b = tile >> 11;
  int vbase = (tile & 2047) << 4;
  int x = vbase >> 10, y = (vbase >> 5) & 31, z0 = vbase & 31;
  int n = lane & 15, khalf = lane >> 4;
  v8f acc0 = {0.f,0.f,0.f,0.f,0.f,0.f,0.f,0.f};
  v8f acc1 = {0.f,0.f,0.f,0.f,0.f,0.f,0.f,0.f};
  const v16h* wAv = (const v16h*)wA;
  const _Float16* xb = xt + (size_t)b*VV*CC + khalf*16;
  int tap = 0;
  #pragma unroll
  for (int dx=-1; dx<=1; ++dx)
  #pragma unroll
  for (int dy=-1; dy<=1; ++dy)
  #pragma unroll
  for (int dz=-1; dz<=1; ++dz){
    int xx = x+dx, yy = y+dy, zz = z0+n+dz;
    bool inb = ((unsigned)xx < 32u) && ((unsigned)yy < 32u) && ((unsigned)zz < 32u);
    int xc = min(max(xx,0),31), yc = min(max(yy,0),31), zc = min(max(zz,0),31);
    int vox = (xc*32+yc)*32+zc;
    v16h bf = *(const v16h*)(xb + (size_t)vox*CC);    // 16 consecutive ci = whole B-frag
    _Float16 mk = inb ? (_Float16)1.f : (_Float16)0.f;
    bf = bf * mk;                                     // branchless zero-pad
    v16h a0 = wAv[(tap*2+0)*32 + lane];
    v16h a1 = wAv[(tap*2+1)*32 + lane];
    acc0 = __builtin_amdgcn_wmma_f32_16x16x32_f16(false,a0,false,bf,(short)0,acc0,false,false);
    acc1 = __builtin_amdgcn_wmma_f32_16x16x32_f16(false,a1,false,bf,(short)0,acc1,false,false);
    ++tap;
  }
  int mrow = (lane & 16) ? 8 : 0;                 // C/D layout: M = e (+8 for upper lanes)
  int ob = b*CC*VV + vbase + n;
  #pragma unroll
  for (int e=0; e<8; ++e){
    int co0 = mrow + e, co1 = co0 + 16;
    out[ob + co0*VV] = acc0[e] + bias[co0];
    out[ob + co1*VV] = acc1[e] + bias[co1];
  }
}

// ---------------- 1x1 point conv on WMMA (channel-last f16 input) ----------------
__global__ void k_pconv(const _Float16* __restrict__ ft, const unsigned int* __restrict__ wA,
                        const float* __restrict__ bias, float* __restrict__ out){
  int lane = threadIdx.x & 31, wave = threadIdx.x >> 5;
  int tile = blockIdx.x*8 + wave;                 // 8192 tiles = B * N/16
  int b = tile >> 10;
  int pbase = (tile & 1023) << 4;
  int n = lane & 15, khalf = lane >> 4;
  v8f acc0 = {0.f,0.f,0.f,0.f,0.f,0.f,0.f,0.f};
  v8f acc1 = {0.f,0.f,0.f,0.f,0.f,0.f,0.f,0.f};
  v16h bf = *(const v16h*)(ft + ((size_t)b*NN + pbase + n)*CC + khalf*16);
  const v16h* wAv = (const v16h*)wA;
  v16h a0 = wAv[lane];
  v16h a1 = wAv[32 + lane];
  acc0 = __builtin_amdgcn_wmma_f32_16x16x32_f16(false,a0,false,bf,(short)0,acc0,false,false);
  acc1 = __builtin_amdgcn_wmma_f32_16x16x32_f16(false,a1,false,bf,(short)0,acc1,false,false);
  int mrow = (lane & 16) ? 8 : 0;
  int ob = b*CC*NN + pbase + n;
  #pragma unroll
  for (int e=0; e<8; ++e){
    int co0 = mrow + e, co1 = co0 + 16;
    out[ob + co0*NN] = acc0[e] + bias[co0];
    out[ob + co1*NN] = acc1[e] + bias[co1];
  }
}

// ---------------- GroupNorm stats: one WG per (b, group) ----------------
__global__ void k_gnstats(const float* __restrict__ x, int S, float* __restrict__ stats){
  int bg = blockIdx.x; int b = bg>>3, g = bg&7;
  int tid = threadIdx.x;
  __shared__ float rs[256], rq[256];
  const float* base = x + (b*CC + g*GC)*S;
  int tot = GC*S;
  float s=0.f, q=0.f;
  for (int i=tid;i<tot;i+=256){ float v = base[i]; s+=v; q+=v*v; }
  rs[tid]=s; rq[tid]=q; __syncthreads();
  for (int st=128;st>0;st>>=1){
    if (tid<st){ rs[tid]+=rs[tid+st]; rq[tid]+=rq[tid+st]; }
    __syncthreads();
  }
  if (tid==0){
    float mean = rs[0]/tot;
    float var  = rq[0]/tot - mean*mean;
    stats[bg*2]   = mean;
    stats[bg*2+1] = rsqrtf(fmaxf(var,0.f)+1e-5f);
  }
}

// ---------------- AdaGN apply (+optional SiLU) ----------------
__global__ void k_apply(const float* __restrict__ x, const float* __restrict__ stats,
                        const float* __restrict__ m, const float* __restrict__ gw,
                        const float* __restrict__ gb, float* __restrict__ out,
                        int S, int dosilu){
  long i = (long)blockIdx.x*256 + threadIdx.x;
  long tot = (long)BB*CC*S;
  if (i >= tot) return;
  int b = (int)(i / ((long)CC*S));
  int c = (int)((i / S) % CC);
  int bg = b*8 + (c>>2);
  float y = (x[i]-stats[bg*2])*stats[bg*2+1]*gw[c] + gb[c];
  y = y*m[b*64+c] + m[b*64+32+c];
  if (dosilu) y = y * sigmoidf(y);
  out[i] = y;
}

// ---------------- SE: channel means + gate ----------------
__global__ void k_chmean(const float* __restrict__ x, float* __restrict__ chm){
  int bc = blockIdx.x, tid = threadIdx.x;
  __shared__ float r[256];
  const float* base = x + bc*VV;
  float s=0.f;
  for (int i=tid;i<VV;i+=256) s += base[i];
  r[tid]=s; __syncthreads();
  for (int st=128;st>0;st>>=1){ if (tid<st) r[tid]+=r[tid+st]; __syncthreads(); }
  if (tid==0) chm[bc] = r[0]/VV;
}

__global__ void k_se(const float* __restrict__ chm, const float* __restrict__ w1,
                     const float* __restrict__ w2, float* __restrict__ s){
  int t = threadIdx.x;          // 256 = B*C
  int b = t>>5, c = t&31;
  float hid[4];
  #pragma unroll
  for (int j=0;j<4;++j){
    float a=0.f;
    for (int k=0;k<CC;++k) a += chm[b*CC+k]*w1[j*CC+k];
    hid[j] = fmaxf(a,0.f);
  }
  float sv=0.f;
  #pragma unroll
  for (int j=0;j<4;++j) sv += hid[j]*w2[c*4+j];
  s[t] = sigmoidf(sv);
}

// ---------------- final: trilinear devox * SE + point branch AdaGN+SiLU ----------------
__global__ void k_final(const float* __restrict__ px, const float* __restrict__ stats,
                        const float* __restrict__ m, const float* __restrict__ gw,
                        const float* __restrict__ gb, const float* __restrict__ h2,
                        const float* __restrict__ s, const float* __restrict__ nc,
                        float* __restrict__ out){
  long i = (long)blockIdx.x*256 + threadIdx.x;
  if (i >= (long)BB*CC*NN) return;
  int n = (int)(i % NN);
  int c = (int)((i / NN) % CC);
  int b = (int)(i / ((long)CC*NN));
  int bg = b*8 + (c>>2);
  float y = (px[i]-stats[bg*2])*stats[bg*2+1]*gw[c] + gb[c];
  y = y*m[b*64+c] + m[b*64+32+c];
  y = y * sigmoidf(y);
  float cx = nc[(b*3+0)*NN+n], cy = nc[(b*3+1)*NN+n], cz = nc[(b*3+2)*NN+n];
  float lx = floorf(cx), ly = floorf(cy), lz = floorf(cz);
  float fx = cx-lx, fy = cy-ly, fz = cz-lz;
  int ix0=(int)lx, iy0=(int)ly, iz0=(int)lz;
  int ix1=min(ix0+1,31), iy1=min(iy0+1,31), iz1=min(iz0+1,31);
  const float* hb = h2 + (b*CC+c)*VV;
  float acc =
      hb[(ix0*32+iy0)*32+iz0]*(1.f-fx)*(1.f-fy)*(1.f-fz)
    + hb[(ix0*32+iy0)*32+iz1]*(1.f-fx)*(1.f-fy)*fz
    + hb[(ix0*32+iy1)*32+iz0]*(1.f-fx)*fy*(1.f-fz)
    + hb[(ix0*32+iy1)*32+iz1]*(1.f-fx)*fy*fz
    + hb[(ix1*32+iy0)*32+iz0]*fx*(1.f-fy)*(1.f-fz)
    + hb[(ix1*32+iy0)*32+iz1]*fx*(1.f-fy)*fz
    + hb[(ix1*32+iy1)*32+iz0]*fx*fy*(1.f-fz)
    + hb[(ix1*32+iy1)*32+iz1]*fx*fy*fz;
  out[i] = acc*s[b*CC+c] + y;
}

extern "C" void kernel_launch(void* const* d_in, const int* in_sizes, int n_in,
                              void* d_out, int out_size, void* d_ws, size_t ws_size,
                              hipStream_t stream){
  const float* features = (const float*)d_in[0];
  const float* coords   = (const float*)d_in[1];
  const float* cond     = (const float*)d_in[2];
  const float* conv1_w  = (const float*)d_in[3];
  const float* conv1_b  = (const float*)d_in[4];
  const float* gn1_g    = (const float*)d_in[5];
  const float* gn1_b    = (const float*)d_in[6];
  const float* mod1_w   = (const float*)d_in[7];
  const float* mod1_b   = (const float*)d_in[8];
  const float* conv2_w  = (const float*)d_in[9];
  const float* conv2_b  = (const float*)d_in[10];
  const float* gn2_g    = (const float*)d_in[11];
  const float* gn2_b    = (const float*)d_in[12];
  const float* mod2_w   = (const float*)d_in[13];
  const float* mod2_b   = (const float*)d_in[14];
  const float* se_w1    = (const float*)d_in[15];
  const float* se_w2    = (const float*)d_in[16];
  const float* pc_w     = (const float*)d_in[17];
  const float* pc_b     = (const float*)d_in[18];
  const float* gnp_g    = (const float*)d_in[19];
  const float* gnp_b    = (const float*)d_in[20];
  const float* modp_w   = (const float*)d_in[21];
  const float* modp_b   = (const float*)d_in[22];
  float* out = (float*)d_out;

  float* ws = (float*)d_ws;
  float* nc   = ws;                          // 393216
  int*   vidx = (int*)(ws + 393216);         // 131072
  float* cnt  = ws + 524288;                 // 262144
  float* VA   = ws + 786432;                 // 8388608 (vox, later v2)
  float* VB   = ws + 9175040;                // 8388608 (v1)
  float* VC   = ws + 17563648;               // 8388608 (h1, later h2)
  float* px   = ws + 25952256;               // 4194304
  float* m    = ws + 30146560;               // 1536 (m1,m2,mp)
  float* st1  = ws + 30148096;               // 128
  float* st2  = ws + 30148224;               // 128
  float* stp  = ws + 30148352;               // 128
  float* chm  = ws + 30148480;               // 256
  float* sbuf = ws + 30148736;               // 256
  unsigned int* wA1 = (unsigned int*)(ws + 30148992);  // 13824 dwords
  unsigned int* wA2 = (unsigned int*)(ws + 30162816);  // 13824 dwords
  unsigned int* wAp = (unsigned int*)(ws + 30176640);  // 512 dwords
  _Float16* xt = (_Float16*)(ws + 30177152);           // 8388608 halves (16MB)
  float* m1 = m; float* m2 = m + 512; float* mp = m + 1024;

  k_prep_w <<<110,   256, 0, stream>>>(conv1_w, conv2_w, pc_w, wA1, wA2, wAp);
  k_coord  <<<BB,    256, 0, stream>>>(coords, nc, vidx);
  k_zero   <<<4096,  256, 0, stream>>>(cnt, (long)(262144 + 8388608));  // cnt+vox contiguous
  k_scatter<<<16384, 256, 0, stream>>>(features, vidx, VA, cnt);
  k_divide <<<32768, 256, 0, stream>>>(VA, cnt);
  k_mod    <<<BB,    256, 0, stream>>>(cond, mod1_w, mod1_b, mod2_w, mod2_b, modp_w, modp_b, m);

  k_tohalf <<<1024,  256, 0, stream>>>(VA, xt, VV);               // vox -> f16 channel-last
  k_conv   <<<2048,  256, 0, stream>>>(xt, wA1, conv1_b, VB);
  k_gnstats<<<64,    256, 0, stream>>>(VB, VV, st1);
  k_apply  <<<32768, 256, 0, stream>>>(VB, st1, m1, gn1_g, gn1_b, VC, VV, 1);

  k_tohalf <<<1024,  256, 0, stream>>>(VC, xt, VV);               // h1 -> f16 channel-last
  k_conv   <<<2048,  256, 0, stream>>>(xt, wA2, conv2_b, VA);
  k_gnstats<<<64,    256, 0, stream>>>(VA, VV, st2);
  k_apply  <<<32768, 256, 0, stream>>>(VA, st2, m2, gn2_g, gn2_b, VC, VV, 0);

  k_chmean <<<256,   256, 0, stream>>>(VC, chm);
  k_se     <<<1,     256, 0, stream>>>(chm, se_w1, se_w2, sbuf);

  k_tohalf <<<512,   256, 0, stream>>>(features, xt, NN);         // features -> f16 ch-last
  k_pconv  <<<1024,  256, 0, stream>>>(xt, wAp, pc_b, px);
  k_gnstats<<<64,    256, 0, stream>>>(px, NN, stp);
  k_final  <<<16384, 256, 0, stream>>>(px, stp, mp, gnp_g, gnp_b, VC, sbuf, nc, out);
  (void)in_sizes; (void)n_in; (void)out_size; (void)ws_size;
}
